// SiteClassifierGraph_29557964931566
// MI455X (gfx1250) — matrix operand order repaired and checked by reading, not compiled, not run on hardware
//
#include <hip/hip_runtime.h>
#include <hip/hip_bf16.h>

typedef __attribute__((ext_vector_type(16))) __bf16 v16bf;
typedef __attribute__((ext_vector_type(4)))  __bf16 v4bf;
typedef __attribute__((ext_vector_type(8)))  float  v8f;
typedef __attribute__((ext_vector_type(4)))  float  v4f;

#define N_NODES 16384
#define NROI    256
#define NGRAPH  64
#define NEDGE   524288
#define LRELU   0.01f
#define BN_EPS  1e-5f

#if defined(__HIP_DEVICE_COMPILE__) && __has_builtin(__builtin_amdgcn_global_load_async_to_lds_b128) && __has_builtin(__builtin_amdgcn_s_wait_asynccnt)
#define HAVE_ASYNC_LDS 1
typedef int b128_t __attribute__((vector_size(16)));          // 4 x i32
typedef __attribute__((address_space(1))) b128_t* gb128_p;    // global ptr
typedef __attribute__((address_space(3))) b128_t* lb128_p;    // LDS ptr
#else
#define HAVE_ASYNC_LDS 0
#endif

// ---------------------------------------------------------------- utilities
__global__ void zero_kernel(float* __restrict__ p, int n) {
  int i = blockIdx.x * blockDim.x + threadIdx.x;
  if (i < n) p[i] = 0.0f;
}

__global__ void deg_kernel(const float* __restrict__ ea, const int* __restrict__ src,
                           float* __restrict__ deg, int E) {
  int e = blockIdx.x * blockDim.x + threadIdx.x;
  if (e < E) atomicAdd(&deg[src[e]], ea[e]);
}

__global__ void dis_kernel(const float* __restrict__ deg, float* __restrict__ dis, int n) {
  int i = blockIdx.x * blockDim.x + threadIdx.x;
  if (i < n) { float d = deg[i]; dis[i] = d > 0.f ? rsqrtf(d) : 0.f; }
}

__global__ void wn_kernel(const float* __restrict__ ea, const int* __restrict__ src,
                          const int* __restrict__ dst, const float* __restrict__ dis,
                          float* __restrict__ wn, int E) {
  int e = blockIdx.x * blockDim.x + threadIdx.x;
  if (e < E) wn[e] = -(dis[src[e]] * ea[e] * dis[dst[e]]);
}

// one block per edge, blockDim == F (64..256): scatter wn[e]*x[src] into tx1[dst]
__global__ void spmm_kernel(const float* __restrict__ x, const float* __restrict__ wn,
                            const int* __restrict__ src, const int* __restrict__ dst,
                            float* __restrict__ tx1, int F) {
  int e = blockIdx.x;
  float w = wn[e];
  int s = src[e], d = dst[e];
  int f = threadIdx.x;
  atomicAdd(&tx1[d * F + f], w * x[s * F + f]);
}

// ---------------------------------------------------------------- WMMA GEMM
// H[N,F] = A0[N,K]@W0[K,F] + A1[N,K]@W1[K,F] + bias.
// One wave per 16(M) x 32(N) output strip; A fragment reused across 2 N-tiles.
// sAx stores A in contiguous k-order so each lane's fragment is one
// contiguous 32B run (2x ds_load_b128); sBt is n-major for the same reason.
__global__ __launch_bounds__(32) void gemm2_wmma(
    const float* __restrict__ A0, const float* __restrict__ A1,
    const float* __restrict__ W0, const float* __restrict__ W1,
    const float* __restrict__ bias, float* __restrict__ H,
    int K, int F) {
  __shared__ __align__(32) __bf16 sAx[16][32];   // [m][k]
  __shared__ __align__(32) __bf16 sBt[32][32];   // [n][k]

  const int lane  = threadIdx.x;   // 0..31 (wave32)
  const int lhalf = lane >> 4;     // 0 or 1
  const int lrow  = lane & 15;
  const int m0 = blockIdx.x * 16;
  const int n0 = blockIdx.y * 32;  // two 16-wide N-tiles per wave

  v8f acc0 = {0.f, 0.f, 0.f, 0.f, 0.f, 0.f, 0.f, 0.f};
  v8f acc1 = {0.f, 0.f, 0.f, 0.f, 0.f, 0.f, 0.f, 0.f};

  #pragma unroll
  for (int p = 0; p < 2; ++p) {
    const float* A = (p == 0) ? A0 : A1;
    const float* W = (p == 0) ? W0 : W1;
    for (int k0 = 0; k0 < K; k0 += 32) {
      // A tile: 16x32 f32, 4x float4 per lane (global_load_b128, coalesced)
      #pragma unroll
      for (int q = 0; q < 4; ++q) {
        int flat = q * 128 + lane * 4;
        int row = flat >> 5, kc = flat & 31;
        v4f av = *(const v4f*)(A + (m0 + row) * K + k0 + kc);
        // position: element k -> k (contiguous, matches sBt k-order);
        // aligned 4-chunks stay contiguous -> single 8B LDS store
        int pos = kc;
        v4bf bv;
        bv[0] = (__bf16)av[0]; bv[1] = (__bf16)av[1];
        bv[2] = (__bf16)av[2]; bv[3] = (__bf16)av[3];
        *(v4bf*)&sAx[row][pos] = bv;
      }
      // W tile: 32x32 f32, 8x float4 per lane; transposed (n-major) LDS store
      #pragma unroll
      for (int q = 0; q < 8; ++q) {
        int flat = q * 128 + lane * 4;
        int kr = flat >> 5, nc = flat & 31;
        v4f wv = *(const v4f*)(W + (k0 + kr) * F + n0 + nc);
        sBt[nc + 0][kr] = (__bf16)wv[0];
        sBt[nc + 1][kr] = (__bf16)wv[1];
        sBt[nc + 2][kr] = (__bf16)wv[2];
        sBt[nc + 3][kr] = (__bf16)wv[3];
      }
      __syncthreads();   // single-wave block: S_NOP + ds waitcnt only

      // fragments: one contiguous 32B vector load each (2x ds_load_b128)
      v16bf a  = *(const v16bf*)&sAx[lrow][lhalf << 4];
      v16bf b0 = *(const v16bf*)&sBt[lrow][lhalf << 4];
      v16bf b1 = *(const v16bf*)&sBt[16 + lrow][lhalf << 4];

      acc0 = __builtin_amdgcn_wmma_f32_16x16x32_bf16(
          false, a, false, b0, (short)0, acc0, false, false);
      acc1 = __builtin_amdgcn_wmma_f32_16x16x32_bf16(
          false, a, false, b1, (short)0, acc1, false, false);
      __syncthreads();
    }
  }

  float bv0 = bias[n0 + lrow];
  float bv1 = bias[n0 + 16 + lrow];
  #pragma unroll
  for (int r = 0; r < 8; ++r) {
    int row = m0 + (lhalf << 3) + r;         // C layout: VGPR r -> M = r + 8*lhalf
    H[row * F + n0 + lrow]      = acc0[r] + bv0;
    H[row * F + n0 + 16 + lrow] = acc1[r] + bv1;
  }
}

// ---------------------------------------------------------------- batch norm
__global__ __launch_bounds__(256) void bn_stats_kernel(const float* __restrict__ H,
                                                       int n, int F, float* __restrict__ stats) {
  int c = blockIdx.x;
  int t = threadIdx.x;
  __shared__ float ss[256], sq[256];
  float s = 0.f, q = 0.f;
  for (int i = t; i < n; i += 256) {
    float v = H[i * F + c];
    s += v; q += v * v;
  }
  ss[t] = s; sq[t] = q;
  __syncthreads();
  for (int o = 128; o > 0; o >>= 1) {
    if (t < o) { ss[t] += ss[t + o]; sq[t] += sq[t + o]; }
    __syncthreads();
  }
  if (t == 0) {
    float m   = ss[0] / n;
    float var = sq[0] / n - m * m;
    stats[c]     = m;
    stats[F + c] = rsqrtf(var + BN_EPS);
  }
}

__global__ void bn_lrelu_kernel(const float* __restrict__ H, const float* __restrict__ stats,
                                const float* __restrict__ gamma, const float* __restrict__ beta,
                                float* __restrict__ X, int total, int F) {
  int i = blockIdx.x * blockDim.x + threadIdx.x;
  if (i < total) {
    int c = i % F;
    float v = (H[i] - stats[c]) * stats[F + c] * gamma[c] + beta[c];
    X[i] = v > 0.f ? v : LRELU * v;
  }
}

// ---------------------------------------------------------------- pooling (F=32)
__global__ __launch_bounds__(256) void pool_kernel(const float* __restrict__ x,
                                                   float* __restrict__ xx) {
  int g = blockIdx.x;          // 64 graphs
  int t = threadIdx.x;
  int f = t & 31, rg = t >> 5; // 8 row-groups
  const float* base = x + (g * NROI) * 32;
  float mx = -3.402823466e38f, sm = 0.f;
  for (int r = rg; r < NROI; r += 8) {
    float v = base[r * 32 + f];
    mx = fmaxf(mx, v); sm += v;
  }
  __shared__ float smax[256], ssum[256];
  smax[t] = mx; ssum[t] = sm;
  __syncthreads();
  if (rg == 0) {
    for (int i = 1; i < 8; ++i) { mx = fmaxf(mx, smax[(i << 5) | f]); sm += ssum[(i << 5) | f]; }
    xx[g * 64 + f]      = mx;                // max pool
    xx[g * 64 + 32 + f] = sm * (1.f / NROI); // mean pool
  }
}

// ---------------------------------------------------------------- MLP (single block)
__global__ __launch_bounds__(256) void mlp_kernel(
    const float* __restrict__ xx,
    const float* __restrict__ W1, const float* __restrict__ b1,
    const float* __restrict__ g1, const float* __restrict__ be1,
    const float* __restrict__ W2, const float* __restrict__ b2,
    const float* __restrict__ g2, const float* __restrict__ be2,
    const float* __restrict__ W3, const float* __restrict__ b3,
    float* __restrict__ out) {
  __shared__ float s0[64 * 64];
  __shared__ float s1[64 * 32];
  __shared__ float s2[64 * 16];
  __shared__ float mcol[32], rcol[32];
  int t = threadIdx.x;

#if HAVE_ASYNC_LDS
  // gfx1250 async global->LDS staging of xx (4096 floats = 4x b128 per thread),
  // tracked with ASYNCcnt.
  #pragma unroll
  for (int q = 0; q < 4; ++q) {
    int idx = (q * 256 + t) * 4;
    __builtin_amdgcn_global_load_async_to_lds_b128(
        (gb128_p)(xx + idx), (lb128_p)(s0 + idx), 0, 0);
  }
  __builtin_amdgcn_s_wait_asynccnt(0);
  __syncthreads();
#else
  for (int i = t; i < 64 * 64; i += 256) s0[i] = xx[i];
  __syncthreads();
#endif

  // layer 1: [64x64] @ [64x32]
  for (int i = t; i < 64 * 32; i += 256) {
    int r = i >> 5, c = i & 31;
    float acc = b1[c];
    for (int k = 0; k < 64; ++k) acc += s0[r * 64 + k] * W1[k * 32 + c];
    s1[i] = acc;
  }
  __syncthreads();
  if (t < 32) {
    float s = 0.f, q = 0.f;
    for (int r = 0; r < 64; ++r) { float v = s1[r * 32 + t]; s += v; q += v * v; }
    float m = s * (1.f / 64.f);
    mcol[t] = m; rcol[t] = rsqrtf(q * (1.f / 64.f) - m * m + BN_EPS);
  }
  __syncthreads();
  for (int i = t; i < 64 * 32; i += 256) {
    int c = i & 31;
    float v = (s1[i] - mcol[c]) * rcol[c] * g1[c] + be1[c];
    s1[i] = v > 0.f ? v : LRELU * v;
  }
  __syncthreads();

  // layer 2: [64x32] @ [32x16]
  for (int i = t; i < 64 * 16; i += 256) {
    int r = i >> 4, c = i & 15;
    float acc = b2[c];
    for (int k = 0; k < 32; ++k) acc += s1[r * 32 + k] * W2[k * 16 + c];
    s2[i] = acc;
  }
  __syncthreads();
  if (t < 16) {
    float s = 0.f, q = 0.f;
    for (int r = 0; r < 64; ++r) { float v = s2[r * 16 + t]; s += v; q += v * v; }
    float m = s * (1.f / 64.f);
    mcol[t] = m; rcol[t] = rsqrtf(q * (1.f / 64.f) - m * m + BN_EPS);
  }
  __syncthreads();
  for (int i = t; i < 64 * 16; i += 256) {
    int c = i & 15;
    float v = (s2[i] - mcol[c]) * rcol[c] * g2[c] + be2[c];
    s2[i] = v > 0.f ? v : LRELU * v;
  }
  __syncthreads();

  // layer 3: [64x16] @ [16x4] -> out
  for (int i = t; i < 64 * 4; i += 256) {
    int r = i >> 2, c = i & 3;
    float acc = b3[c];
    for (int k = 0; k < 16; ++k) acc += s2[r * 16 + k] * W3[k * 4 + c];
    out[i] = acc;
  }
}

// ---------------------------------------------------------------- launcher
extern "C" void kernel_launch(void* const* d_in, const int* in_sizes, int n_in,
                              void* d_out, int out_size, void* d_ws, size_t ws_size,
                              hipStream_t stream) {
  (void)in_sizes; (void)n_in; (void)out_size; (void)ws_size;

  // dict order: x, edge_attr, params{conv[4]{W0,W1,b,gamma,beta}, mlp[3]{W,b,[gamma,beta]}},
  //             edge_index, batch_idx, nroi
  const float* x0 = (const float*)d_in[0];
  const float* ea = (const float*)d_in[1];
  const float *cW0[4], *cW1[4], *cb[4], *cg[4], *cbe[4];
  int idx = 2;
  for (int l = 0; l < 4; ++l) {
    cW0[l] = (const float*)d_in[idx++];
    cW1[l] = (const float*)d_in[idx++];
    cb[l]  = (const float*)d_in[idx++];
    cg[l]  = (const float*)d_in[idx++];
    cbe[l] = (const float*)d_in[idx++];
  }
  const float *mW[3], *mb[3], *mg[2], *mbe[2];
  for (int l = 0; l < 3; ++l) {
    mW[l] = (const float*)d_in[idx++];
    mb[l] = (const float*)d_in[idx++];
    if (l < 2) { mg[l] = (const float*)d_in[idx++]; mbe[l] = (const float*)d_in[idx++]; }
  }
  const int* ei = (const int*)d_in[idx++];  // [2, E]
  idx++;                                    // batch_idx (structure is implicit)
  idx++;                                    // nroi scalar
  const int* src = ei;
  const int* dst = ei + NEDGE;

  // workspace layout (floats)
  float* ws    = (float*)d_ws;
  float* deg   = ws; ws += N_NODES;
  float* dis   = ws; ws += N_NODES;
  float* wn    = ws; ws += NEDGE;
  float* stats = ws; ws += 2 * 256;
  float* xx    = ws; ws += NGRAPH * 64;
  float* xA    = ws; ws += (size_t)N_NODES * 256;
  float* xB    = ws; ws += (size_t)N_NODES * 256;
  float* tx1   = ws; ws += (size_t)N_NODES * 256;
  float* h     = ws; ws += (size_t)N_NODES * 256;

  // graph normalization: deg -> dis -> wn
  zero_kernel<<<(N_NODES + 255) / 256, 256, 0, stream>>>(deg, N_NODES);
  deg_kernel<<<(NEDGE + 255) / 256, 256, 0, stream>>>(ea, src, deg, NEDGE);
  dis_kernel<<<(N_NODES + 255) / 256, 256, 0, stream>>>(deg, dis, N_NODES);
  wn_kernel<<<(NEDGE + 255) / 256, 256, 0, stream>>>(ea, src, dst, dis, wn, NEDGE);

  const int dims_in[4]  = {256, 256, 128, 64};
  const int dims_out[4] = {256, 128, 64, 32};
  const float* xin = x0;
  float* bufs[2] = {xA, xB};
  for (int l = 0; l < 4; ++l) {
    int K = dims_in[l], F = dims_out[l];
    int nK = N_NODES * K;
    zero_kernel<<<(nK + 255) / 256, 256, 0, stream>>>(tx1, nK);
    spmm_kernel<<<NEDGE, K, 0, stream>>>(xin, wn, src, dst, tx1, K);
    gemm2_wmma<<<dim3(N_NODES / 16, F / 32), 32, 0, stream>>>(
        xin, tx1, cW0[l], cW1[l], cb[l], h, K, F);
    bn_stats_kernel<<<F, 256, 0, stream>>>(h, N_NODES, F, stats);
    float* xout = bufs[l & 1];
    bn_lrelu_kernel<<<(N_NODES * F + 255) / 256, 256, 0, stream>>>(
        h, stats, cg[l], cbe[l], xout, N_NODES * F, F);
    xin = xout;
  }

  pool_kernel<<<NGRAPH, 256, 0, stream>>>(xin, xx);
  mlp_kernel<<<1, 256, 0, stream>>>(xx,
      mW[0], mb[0], mg[0], mbe[0],
      mW[1], mb[1], mg[1], mbe[1],
      mW[2], mb[2], (float*)d_out);
}
